// eComFormer_18726057411382
// MI455X (gfx1250) — matrix operand order repaired and checked by reading, not compiled, not run on hardware
//
#include <hip/hip_runtime.h>
#include <math.h>
#include <stdint.h>

typedef __bf16 bf16_t;
typedef __attribute__((ext_vector_type(16))) __bf16 v16bf;
typedef __attribute__((ext_vector_type(8)))  float  v8f;

#define NNODES 2048
#define NEDGES 24576
#define NGRAPH 64
#define CDIM   256
#define CHUNK  2048
#define NCHUNK (NEDGES/CHUNK)
#define SQRT3F  1.7320508075688772f
#define SQRT5F  2.23606797749979f
#define SQRT15F 3.872983346207417f

__device__ __forceinline__ float sp_f(float x){ return x > 20.f ? x : log1pf(expf(x)); }
__device__ __forceinline__ float sig_f(float x){ return 1.f/(1.f+expf(-x)); }

__device__ __forceinline__ void cvt8(v16bf& d, int base, float4 a, float4 b){
  d[base+0]=(bf16_t)a.x; d[base+1]=(bf16_t)a.y; d[base+2]=(bf16_t)a.z; d[base+3]=(bf16_t)a.w;
  d[base+4]=(bf16_t)b.x; d[base+5]=(bf16_t)b.y; d[base+6]=(bf16_t)b.z; d[base+7]=(bf16_t)b.w;
}

__device__ __forceinline__ v16bf loadfrag(const float* __restrict__ p){
  float4 q0 = *(const float4*)(p);
  float4 q1 = *(const float4*)(p + 4);
  float4 q2 = *(const float4*)(p + 8);
  float4 q3 = *(const float4*)(p + 12);
  v16bf v; cvt8(v, 0, q0, q1); cvt8(v, 8, q2, q3);
  return v;
}

// ---------------------------------------------------------------------------
// WMMA bf16 GEMM:  C[M,N] = epilogue( A[M,K] @ W[N,K]^T )
// REQUIREMENTS (guaranteed by all call sites):
//   M % 64 == 0 (grid.y = M/64), N % 64 == 0, K % 32 == 0,
//   lda % 4 == 0, ldw % 4 == 0, 16B-aligned base pointers.
// Wave tile 16x64 (one A fragment, 4 accumulators); block = 8 waves = 64x128.
// Inner loop is completely branch-free (every active wave has 4 full tiles);
// idle waves (N==64 case) exit via a scalar-uniform early return.
// ---------------------------------------------------------------------------
__global__ __launch_bounds__(256) void wmma_gemm(
  const float* __restrict__ A, int lda,
  const float* __restrict__ W, int ldw, const float* __restrict__ bias,
  const float* __restrict__ add1, const int* __restrict__ add1i, int add1ld,
  const float* __restrict__ add2, const int* __restrict__ add2i, int add2ld,
  const float* __restrict__ bng, const float* __restrict__ bnb,
  const float* __restrict__ bnrm, const float* __restrict__ bnrv,
  const float* __restrict__ res, int resld, int resafter,
  int act, float* __restrict__ Cout, int ldc, int M, int N, int K)
{
  const int lane = threadIdx.x & 31;
  const int wave = threadIdx.x >> 5;
  const int m0 = blockIdx.y * 64  + (wave & 3) * 16;
  const int n0 = blockIdx.x * 128 + (wave >> 2) * 64;
  const int lr = lane & 15;
  const int hi = lane >> 4;

  // wave-uniform early exit (only the N==64 call sites have idle waves)
  if (__builtin_amdgcn_readfirstlane(n0) >= N) return;

  const float* Ap  = A + (long)(m0 + lr) * lda;
  const float* Wp0 = W + (long)(n0 +  0 + lr) * ldw;
  const float* Wp1 = W + (long)(n0 + 16 + lr) * ldw;
  const float* Wp2 = W + (long)(n0 + 32 + lr) * ldw;
  const float* Wp3 = W + (long)(n0 + 48 + lr) * ldw;

  v8f acc0, acc1, acc2, acc3;
  #pragma unroll
  for (int r = 0; r < 8; ++r) { acc0[r]=0.f; acc1[r]=0.f; acc2[r]=0.f; acc3[r]=0.f; }

  for (int k0 = 0; k0 < K; k0 += 32) {
    // A fragment: lane = row; elems 0..15 -> kbA+0..15
    const int kbA = k0 + hi * 16;
    float4 a0 = *(const float4*)(Ap + kbA);
    float4 a1 = *(const float4*)(Ap + kbA + 4);
    float4 a2 = *(const float4*)(Ap + kbA + 8);
    float4 a3 = *(const float4*)(Ap + kbA + 12);
    v16bf af;
    cvt8(af, 0, a0, a1);
    cvt8(af, 8, a2, a3);
    // B fragments: lane = column; elems 0..15 -> kbB+0..15
    const int kbB = k0 + hi * 16;
    v16bf b0 = loadfrag(Wp0 + kbB);
    v16bf b1 = loadfrag(Wp1 + kbB);
    v16bf b2 = loadfrag(Wp2 + kbB);
    v16bf b3 = loadfrag(Wp3 + kbB);
    acc0 = __builtin_amdgcn_wmma_f32_16x16x32_bf16(false, af, false, b0, (short)0, acc0, false, false);
    acc1 = __builtin_amdgcn_wmma_f32_16x16x32_bf16(false, af, false, b1, (short)0, acc1, false, false);
    acc2 = __builtin_amdgcn_wmma_f32_16x16x32_bf16(false, af, false, b2, (short)0, acc2, false, false);
    acc3 = __builtin_amdgcn_wmma_f32_16x16x32_bf16(false, af, false, b3, (short)0, acc3, false, false);
  }

  // epilogue: C/D layout: VGPR r -> row m0 + r + hi*8, lane%16 -> column
  v8f accs[4] = {acc0, acc1, acc2, acc3};
  #pragma unroll
  for (int t = 0; t < 4; ++t) {
    const int col = n0 + t * 16 + lr;
    const float bia = bias ? bias[col] : 0.f;
    float sc = 1.f, sh = 0.f;
    const bool hasbn = (bng != nullptr);
    if (hasbn) {
      float inv = rsqrtf(bnrv[col] + 1e-5f);
      sc = inv * bng[col];
      sh = bnb[col] - bnrm[col] * sc;
    }
    #pragma unroll
    for (int r = 0; r < 8; ++r) {
      const int row = m0 + r + hi * 8;
      float v = accs[t][r] + bia;
      if (add1) v += add1[(long)add1i[row] * add1ld + col];
      if (add2) v += add2[(long)add2i[row] * add2ld + col];
      if (hasbn) v = v * sc + sh;
      if (res && !resafter) v += res[(long)row * resld + col];
      if (act == 1) v = sp_f(v);
      else if (act == 2) v = v * sig_f(v);
      if (res && resafter) v += res[(long)row * resld + col];
      Cout[(long)row * ldc + col] = v;
    }
  }
}

// zero-pad rows: dst[r, 0:dw] = (c < sw) ? src[r, c] : 0
__global__ void k_pad(const float* __restrict__ src, int sw,
                      float* __restrict__ dst, int dw)
{
  int r = blockIdx.x, c = threadIdx.x;
  dst[(long)r * dw + c] = (c < sw) ? src[(long)r * sw + c] : 0.f;
}

// ---------------------------------------------------------------------------
// Edge prep: RBF features, spherical harmonics (l=1,2), src-degree counts
// ---------------------------------------------------------------------------
__global__ __launch_bounds__(256) void k_edge_prep(
  const float* __restrict__ vec, const int* __restrict__ srcI,
  float* __restrict__ rbf, float* __restrict__ s12, float* __restrict__ deg)
{
  int e = blockIdx.x, c = threadIdx.x;
  float x = vec[e*3+0], y = vec[e*3+1], z = vec[e*3+2];
  float nrm = sqrtf(x*x + y*y + z*z);
  float d = -0.75f / nrm;
  float center = -4.f + 4.f * (float)c / 255.f;
  float t = d - center;
  rbf[(long)e * CDIM + c] = expf(-63.75f * t * t);
  if (c == 0) {
    float inv = 1.f / nrm;
    float vx = x*inv, vy = y*inv, vz = z*inv;
    s12[e*8+0] = SQRT3F*vx;  s12[e*8+1] = SQRT3F*vy;  s12[e*8+2] = SQRT3F*vz;
    s12[e*8+3] = SQRT15F*vx*vz;
    s12[e*8+4] = SQRT15F*vx*vy;
    s12[e*8+5] = SQRT5F*(vy*vy - 0.5f*(vx*vx + vz*vz));
    s12[e*8+6] = SQRT15F*vy*vz;
    s12[e*8+7] = 0.5f*SQRT15F*(vz*vz - vx*vx);
    atomicAdd(&deg[srcI[e]], 1.f);
  }
}

// gated message scatter: agg[dst] += msg * sigmoid(bn(q[dst]*key/sqrt(C)))
__global__ __launch_bounds__(256) void k_attn_edge(
  const float* __restrict__ q, const float* __restrict__ key,
  const float* __restrict__ msg, const int* __restrict__ dstI,
  const float* __restrict__ g, const float* __restrict__ b,
  const float* __restrict__ rm, const float* __restrict__ rv,
  float* __restrict__ agg)
{
  int e = blockIdx.x, c = threadIdx.x;
  int d = dstI[e];
  float alpha = q[(long)d*CDIM + c] * key[(long)e*CDIM + c] * (1.f/16.f);
  float inv = rsqrtf(rv[c] + 1e-5f);
  float t = (alpha - rm[c]) * inv * g[c] + b[c];
  atomicAdd(&agg[(long)d*CDIM + c], msg[(long)e*CDIM + c] * sig_f(t));
}

// equi einsum #1: contract x0[dst] with chunked w=[E,5120] -> tp scatter on src
__global__ __launch_bounds__(64) void k_einsum1(
  const float* __restrict__ wch, int e0, const int* __restrict__ srcI,
  const int* __restrict__ dstI, const float* __restrict__ x0,
  const float* __restrict__ s12, float* __restrict__ tp_sum)
{
  int ce = blockIdx.x, e = e0 + ce, v = threadIdx.x;
  __shared__ float xj[64];
  __shared__ float sph[8];
  int d = dstI[e], s = srcI[e];
  xj[v] = x0[d*64 + v];
  if (v < 8) sph[v] = s12[e*8 + v];
  __syncthreads();
  const float* w = wch + (long)ce * 5120;
  float o0 = 0.f;
  #pragma unroll 8
  for (int u = 0; u < 64; ++u) o0 += xj[u] * w[u*64 + v];
  atomicAdd(&tp_sum[s*128 + v], o0);
  if (v < 8) {
    float t1 = 0.f;
    for (int u = 0; u < 64; ++u) t1 += xj[u] * w[4096 + u*8 + v];
    t1 *= (1.f/SQRT3F);
    for (int m = 0; m < 3; ++m) atomicAdd(&tp_sum[s*128 + 64 + v*3 + m], t1*sph[m]);
  } else if (v < 16) {
    int v2 = v - 8;
    float t2 = 0.f;
    for (int u = 0; u < 64; ++u) t2 += xj[u] * w[4608 + u*8 + v2];
    t2 *= (1.f/SQRT5F);
    for (int m = 0; m < 5; ++m) atomicAdd(&tp_sum[s*128 + 88 + v2*5 + m], t2*sph[3+m]);
  }
}

// equi einsum #2: contract h1[dst] (with sph) against chunked w -> o scatter on src
__global__ __launch_bounds__(64) void k_einsum2(
  const float* __restrict__ wch, int e0, const int* __restrict__ srcI,
  const int* __restrict__ dstI, const float* __restrict__ h1,
  const float* __restrict__ s12, float* __restrict__ o_sum)
{
  int ce = blockIdx.x, e = e0 + ce, v = threadIdx.x;
  __shared__ float coef[80];
  int d = dstI[e], s = srcI[e];
  coef[v] = h1[d*128 + v];
  if (v < 8) {
    float c1 = 0.f;
    for (int m = 0; m < 3; ++m) c1 += h1[d*128 + 64 + v*3 + m] * s12[e*8 + m];
    coef[64 + v] = c1 * (1.f/SQRT3F);
    float c2 = 0.f;
    for (int m = 0; m < 5; ++m) c2 += h1[d*128 + 88 + v*5 + m] * s12[e*8 + 3 + m];
    coef[72 + v] = c2 * (1.f/SQRT5F);
  }
  __syncthreads();
  const float* w = wch + (long)ce * 5120;
  float o = 0.f;
  #pragma unroll 8
  for (int u = 0; u < 64; ++u) o += coef[u] * w[u*64 + v];
  #pragma unroll
  for (int u = 0; u < 8; ++u) o += coef[64 + u] * w[4096 + u*64 + v];
  #pragma unroll
  for (int u = 0; u < 8; ++u) o += coef[72 + u] * w[4608 + u*64 + v];
  atomicAdd(&o_sum[s*64 + v], o);
}

__global__ __launch_bounds__(128) void k_fin_h1(
  const float* __restrict__ tp, const float* __restrict__ deg,
  const float* __restrict__ x0, float* __restrict__ h1)
{
  int i = blockIdx.x, j = threadIdx.x;
  float v = tp[i*128 + j] / fmaxf(deg[i], 1.f);
  if (j < 64) v += x0[i*64 + j];
  h1[i*128 + j] = v;
}

__global__ __launch_bounds__(64) void k_fin_z(
  const float* __restrict__ osum, const float* __restrict__ deg,
  const float* __restrict__ g, const float* __restrict__ b,
  const float* __restrict__ rm, const float* __restrict__ rv,
  float* __restrict__ z)
{
  int i = blockIdx.x, j = threadIdx.x;
  float v = osum[i*64 + j] / fmaxf(deg[i], 1.f);
  float inv = rsqrtf(rv[j] + 1e-5f);
  v = (v - rm[j]) * inv * g[j] + b[j];
  z[i*64 + j] = sp_f(v);
}

__global__ __launch_bounds__(256) void k_pool(
  const float* __restrict__ h, const int* __restrict__ batch,
  float* __restrict__ gsum, float* __restrict__ gcnt)
{
  int i = blockIdx.x, c = threadIdx.x;
  int b = batch[i];
  atomicAdd(&gsum[b*CDIM + c], h[(long)i*CDIM + c]);
  if (c == 0) atomicAdd(&gcnt[b], 1.f);
}

__global__ __launch_bounds__(256) void k_gdiv(
  const float* __restrict__ gsum, const float* __restrict__ gcnt,
  float* __restrict__ gbuf)
{
  int g = blockIdx.x, c = threadIdx.x;
  gbuf[g*CDIM + c] = gsum[g*CDIM + c] / fmaxf(gcnt[g], 1.f);
}

__global__ __launch_bounds__(256) void k_out(
  const float* __restrict__ f, const float* __restrict__ w,
  const float* __restrict__ b, float* __restrict__ out)
{
  __shared__ float red[256];
  int g = blockIdx.x, t = threadIdx.x;
  red[t] = f[g*CDIM + t] * w[t];
  __syncthreads();
  for (int s = 128; s > 0; s >>= 1) {
    if (t < s) red[t] += red[t + s];
    __syncthreads();
  }
  if (t == 0) out[g] = red[0] + b[0];
}

// ---------------------------------------------------------------------------
static inline void gemm(hipStream_t st,
  const float* A, int lda,
  const float* W, int ldw, const float* bias,
  const float* a1, const int* a1i, int a1ld,
  const float* a2, const int* a2i, int a2ld,
  const float* g, const float* b, const float* rm, const float* rv,
  const float* res, int resld, int resafter, int act,
  float* C, int ldc, int M, int N, int K)
{
  dim3 gr((N + 127) / 128, M / 64);
  wmma_gemm<<<gr, 256, 0, st>>>(A, lda, W, ldw, bias, a1, a1i, a1ld,
                                a2, a2i, a2ld, g, b, rm, rv,
                                res, resld, resafter, act, C, ldc, M, N, K);
}

extern "C" void kernel_launch(void* const* d_in, const int* in_sizes, int n_in,
                              void* d_out, int out_size, void* d_ws, size_t ws_size,
                              hipStream_t stream)
{
  (void)in_sizes; (void)n_in; (void)out_size; (void)ws_size;
  const float* X    = (const float*)d_in[0];
  const float* EVEC = (const float*)d_in[1];
  const int*   EIDX = (const int*)d_in[2];
  const int*   BATCH= (const int*)d_in[3];
  const int* srcI = EIDX;
  const int* dstI = EIDX + NEDGES;
  auto F = [&](int i) { return (const float*)d_in[i]; };

  // workspace carve-up (floats)
  float* ws = (float*)d_ws;
  size_t off = 0;
  auto alloc = [&](size_t n) { float* p = ws + off; off += n; return p; };
  float* hA     = alloc((size_t)NNODES * CDIM);
  float* hB     = alloc((size_t)NNODES * CDIM);
  float* nT1    = alloc((size_t)NNODES * CDIM);
  float* nT2    = alloc((size_t)NNODES * CDIM);
  float* nT3    = alloc((size_t)NNODES * CDIM);
  float* e_feat = alloc((size_t)NEDGES * CDIM);
  float* eT1    = alloc((size_t)NEDGES * CDIM);
  float* eT2    = alloc((size_t)NEDGES * CDIM);   // wchunk aliases eT2+eT3
  float* eT3    = alloc((size_t)NEDGES * CDIM);
  float* eT4    = alloc((size_t)NEDGES * CDIM);
  float* s12    = alloc((size_t)NEDGES * 8);
  float* deg    = alloc(NNODES);
  float* x0b    = alloc((size_t)NNODES * 64);
  float* h1b    = alloc((size_t)NNODES * 128);
  float* zb     = alloc((size_t)NNODES * 64);
  float* tp_sum = alloc((size_t)NNODES * 128);
  float* o_sum  = alloc((size_t)NNODES * 64);
  float* g_sum  = alloc((size_t)NGRAPH * CDIM);
  float* g_cnt  = alloc(NGRAPH);
  float* g_buf  = alloc((size_t)NGRAPH * CDIM);
  float* f_buf  = alloc((size_t)NGRAPH * CDIM);
  float* Xp     = alloc((size_t)NNODES * 96);   // x zero-padded K: 92 -> 96
  float* Wemb   = alloc((size_t)CDIM * 96);     // emb_w zero-padded
  float* wchunk = eT2;   // 2048*5120 floats fits in eT2+eT3 (2 * 6.29M)

  const float* NUL = nullptr;
  const int*   NUI = nullptr;

  // 1) pad x / emb_w to K=96, then hA = x @ emb_w^T + emb_b  [2048,256]
  k_pad<<<NNODES, 96, 0, stream>>>(X, 92, Xp, 96);
  k_pad<<<CDIM, 96, 0, stream>>>(F(4), 92, Wemb, 96);
  gemm(stream, Xp, 96, Wemb, 96, F(5), NUL,NUI,0, NUL,NUI,0,
       NUL,NUL,NUL,NUL, NUL,0,0, 0, hA, CDIM, NNODES, CDIM, 96);

  // 2) RBF + sph harm + deg, then e_feat = softplus(rbf @ rbf_w^T + b)
  hipMemsetAsync(deg, 0, NNODES*sizeof(float), stream);
  k_edge_prep<<<NEDGES, 256, 0, stream>>>(EVEC, srcI, eT1, s12, deg);
  gemm(stream, eT1, CDIM, F(6), CDIM, F(7), NUL,NUI,0, NUL,NUI,0,
       NUL,NUL,NUL,NUL, NUL,0,0, 1, e_feat, CDIM, NEDGES, CDIM, CDIM);

  // -------- attention conv --------
  auto conv = [&](int pb, const float* hin, float* hout) {
    // K path
    gemm(stream, hin, CDIM, F(pb+2), CDIM, F(pb+3), NUL,NUI,0, NUL,NUI,0,
         NUL,NUL,NUL,NUL, NUL,0,0, 0, nT1, CDIM, NNODES, CDIM, CDIM);            // k
    gemm(stream, nT1, CDIM, F(pb+10)+0,   768, NUL, NUL,NUI,0, NUL,NUI,0,
         NUL,NUL,NUL,NUL, NUL,0,0, 0, nT2, CDIM, NNODES, CDIM, CDIM);            // k@Wa
    gemm(stream, nT1, CDIM, F(pb+10)+256, 768, NUL, NUL,NUI,0, NUL,NUI,0,
         NUL,NUL,NUL,NUL, NUL,0,0, 0, nT3, CDIM, NNODES, CDIM, CDIM);            // k@Wb
    gemm(stream, e_feat, CDIM, F(pb+6), CDIM, F(pb+7), NUL,NUI,0, NUL,NUI,0,
         NUL,NUL,NUL,NUL, NUL,0,0, 0, eT4, CDIM, NEDGES, CDIM, CDIM);            // e
    gemm(stream, eT4, CDIM, F(pb+10)+512, 768, F(pb+11),
         nT2, dstI, CDIM, nT3, srcI, CDIM,
         NUL,NUL,NUL,NUL, NUL,0,0, 2, eT1, CDIM, NEDGES, CDIM, CDIM);            // silu
    gemm(stream, eT1, CDIM, F(pb+12), CDIM, F(pb+13), NUL,NUI,0, NUL,NUI,0,
         NUL,NUL,NUL,NUL, NUL,0,0, 0, eT2, CDIM, NEDGES, CDIM, CDIM);            // key
    // V path (e reused from eT4)
    gemm(stream, hin, CDIM, F(pb+4), CDIM, F(pb+5), NUL,NUI,0, NUL,NUI,0,
         NUL,NUL,NUL,NUL, NUL,0,0, 0, nT1, CDIM, NNODES, CDIM, CDIM);            // v
    gemm(stream, nT1, CDIM, F(pb+14)+0,   768, NUL, NUL,NUI,0, NUL,NUI,0,
         NUL,NUL,NUL,NUL, NUL,0,0, 0, nT2, CDIM, NNODES, CDIM, CDIM);
    gemm(stream, nT1, CDIM, F(pb+14)+256, 768, NUL, NUL,NUI,0, NUL,NUI,0,
         NUL,NUL,NUL,NUL, NUL,0,0, 0, nT3, CDIM, NNODES, CDIM, CDIM);
    gemm(stream, eT4, CDIM, F(pb+14)+512, 768, F(pb+15),
         nT2, dstI, CDIM, nT3, srcI, CDIM,
         NUL,NUL,NUL,NUL, NUL,0,0, 2, eT1, CDIM, NEDGES, CDIM, CDIM);
    gemm(stream, eT1, CDIM, F(pb+16), CDIM, F(pb+17), NUL,NUI,0, NUL,NUI,0,
         NUL,NUL,NUL,NUL, NUL,0,0, 0, eT3, CDIM, NEDGES, CDIM, CDIM);            // msg
    // Q + gated scatter
    gemm(stream, hin, CDIM, F(pb+0), CDIM, F(pb+1), NUL,NUI,0, NUL,NUI,0,
         NUL,NUL,NUL,NUL, NUL,0,0, 0, nT1, CDIM, NNODES, CDIM, CDIM);            // q
    hipMemsetAsync(nT2, 0, (size_t)NNODES*CDIM*sizeof(float), stream);
    k_attn_edge<<<NEDGES, 256, 0, stream>>>(nT1, eT2, eT3, dstI,
                 F(pb+22), F(pb+23), F(pb+24), F(pb+25), nT2);
    // hout = softplus(h + bn(agg @ cat^T + b))
    gemm(stream, nT2, CDIM, F(pb+8), CDIM, F(pb+9), NUL,NUI,0, NUL,NUI,0,
         F(pb+18), F(pb+19), F(pb+20), F(pb+21),
         hin, CDIM, 0, 1, hout, CDIM, NNODES, CDIM, CDIM);
  };

  conv(8, hA, hB);   // att0

  // -------- equivariant update (params base 86): hB -> hA --------
  {
    const int pb = 86;
    gemm(stream, hB, CDIM, F(pb+2), CDIM, F(pb+3), NUL,NUI,0, NUL,NUI,0,
         NUL,NUL,NUL,NUL, NUL,0,0, 0, nT1, CDIM, NNODES, CDIM, CDIM);            // skip
    gemm(stream, hB, CDIM, F(pb+0), CDIM, F(pb+1), NUL,NUI,0, NUL,NUI,0,
         NUL,NUL,NUL,NUL, NUL,0,0, 0, x0b, 64, NNODES, 64, CDIM);                // x0
    gemm(stream, e_feat, CDIM, F(pb+4), CDIM, F(pb+5), NUL,NUI,0, NUL,NUI,0,
         NUL,NUL,NUL,NUL, NUL,0,0, 1, eT1, CDIM, NEDGES, CDIM, CDIM);            // a1
    hipMemsetAsync(tp_sum, 0, (size_t)NNODES*128*sizeof(float), stream);
    for (int c = 0; c < NCHUNK; ++c) {
      gemm(stream, eT1 + (size_t)c*CHUNK*CDIM, CDIM, F(pb+6), CDIM, F(pb+7),
           NUL,NUI,0, NUL,NUI,0, NUL,NUL,NUL,NUL, NUL,0,0, 0,
           wchunk, 5120, CHUNK, 5120, CDIM);                                     // fc1_2
      k_einsum1<<<CHUNK, 64, 0, stream>>>(wchunk, c*CHUNK, srcI, dstI,
                                          x0b, s12, tp_sum);
    }
    k_fin_h1<<<NNODES, 128, 0, stream>>>(tp_sum, deg, x0b, h1b);
    gemm(stream, e_feat, CDIM, F(pb+8), CDIM, F(pb+9), NUL,NUI,0, NUL,NUI,0,
         NUL,NUL,NUL,NUL, NUL,0,0, 1, eT1, CDIM, NEDGES, CDIM, CDIM);            // a2
    hipMemsetAsync(o_sum, 0, (size_t)NNODES*64*sizeof(float), stream);
    for (int c = 0; c < NCHUNK; ++c) {
      gemm(stream, eT1 + (size_t)c*CHUNK*CDIM, CDIM, F(pb+10), CDIM, F(pb+11),
           NUL,NUI,0, NUL,NUI,0, NUL,NUL,NUL,NUL, NUL,0,0, 0,
           wchunk, 5120, CHUNK, 5120, CDIM);                                     // fc2_2
      k_einsum2<<<CHUNK, 64, 0, stream>>>(wchunk, c*CHUNK, srcI, dstI,
                                          h1b, s12, o_sum);
    }
    k_fin_z<<<NNODES, 64, 0, stream>>>(o_sum, deg, F(pb+12), F(pb+13),
                                       F(pb+14), F(pb+15), zb);
    // hA = softplus(z @ node_lin2^T + b) + skip
    gemm(stream, zb, 64, F(pb+16), 64, F(pb+17), NUL,NUI,0, NUL,NUI,0,
         NUL,NUL,NUL,NUL, nT1, CDIM, 1, 1, hA, CDIM, NNODES, CDIM, 64);
  }

  conv(34, hA, hB);  // att1
  conv(60, hB, hA);  // att2

  // -------- readout --------
  hipMemsetAsync(g_sum, 0, (size_t)NGRAPH*CDIM*sizeof(float), stream);
  hipMemsetAsync(g_cnt, 0, NGRAPH*sizeof(float), stream);
  k_pool<<<NNODES, 256, 0, stream>>>(hA, BATCH, g_sum, g_cnt);
  k_gdiv<<<NGRAPH, 256, 0, stream>>>(g_sum, g_cnt, g_buf);
  gemm(stream, g_buf, CDIM, F(104), CDIM, F(105), NUL,NUI,0, NUL,NUI,0,
       NUL,NUL,NUL,NUL, NUL,0,0, 2, f_buf, CDIM, NGRAPH, CDIM, CDIM);           // silu(fc)
  k_out<<<NGRAPH, 256, 0, stream>>>(f_buf, F(106), F(107), (float*)d_out);
}
